// contrastive_loss_85418309583112
// MI455X (gfx1250) — matrix operand order repaired and checked by reading, not compiled, not run on hardware
//
#include <hip/hip_runtime.h>
#include <math.h>

typedef __attribute__((ext_vector_type(4)))  _Float16 v4h;
typedef __attribute__((ext_vector_type(8)))  _Float16 v8h;
typedef __attribute__((ext_vector_type(16))) _Float16 v16h;
typedef __attribute__((ext_vector_type(8)))  float    v8f;
typedef int v4i __attribute__((vector_size(16)));   // matches builtin param type

#define N_ROWS 8192
#define DIM    512
#define MB     64      // output rows per block
#define NB     64      // output cols per j-iteration
#define KC     64      // K chunk resident in LDS
#define KP     72      // padded LDS row stride (halves): 144B = 16B-aligned, bank-rotating
#define JSPLIT 4       // j-range splits across gridDim.y
#define EULER_E 2.71828182845904523536f
#define NMAT   ((size_t)N_ROWS * DIM)

static __device__ __forceinline__ v16h combine16(v8h a, v8h b) {
    v16h r;
#pragma unroll
    for (int i = 0; i < 8; ++i) { r[i] = a[i]; r[i + 8] = b[i]; }
    return r;
}

// 16-byte async global->LDS copy (ASYNCcnt-tracked, gfx1250).
static __device__ __forceinline__ void async_copy16(void* lds, const void* g) {
#if defined(__HIP_DEVICE_COMPILE__)
#if __has_builtin(__builtin_amdgcn_global_load_async_to_lds_b128)
    __builtin_amdgcn_global_load_async_to_lds_b128(
        (__attribute__((address_space(1))) v4i*)g,
        (__attribute__((address_space(3))) v4i*)lds, 0, 0);
#else
    unsigned lofs = (unsigned)(size_t)(__attribute__((address_space(3))) void*)lds;
    asm volatile("global_load_async_to_lds_b128 %0, %1, off"
                 :: "v"(lofs), "v"(g) : "memory");
#endif
#else
    (void)lds; (void)g;
#endif
}

static __device__ __forceinline__ void wait_async0() {
#if defined(__HIP_DEVICE_COMPILE__)
#if __has_builtin(__builtin_amdgcn_s_wait_asynccnt)
    __builtin_amdgcn_s_wait_asynccnt(0);
#else
    asm volatile("s_wait_asynccnt 0" ::: "memory");
#endif
#endif
}

// ---------------------------------------------------------------------------
// Kernel 0: one-time f32 -> f16 (hi, lo) split of both matrices.
// x = hi + lo recovers ~22 mantissa bits across 3 f16 WMMAs (hh + hl + lh).
// ---------------------------------------------------------------------------
__global__ __launch_bounds__(256) void cvt_kernel(
    const float* __restrict__ xi, const float* __restrict__ xj,
    _Float16* __restrict__ xiHi, _Float16* __restrict__ xiLo,
    _Float16* __restrict__ xjHi, _Float16* __restrict__ xjLo)
{
    const size_t i4 = ((size_t)blockIdx.x * 256 + threadIdx.x) * 4;
    const float4 a = *(const float4*)(xi + i4);
    const float4 b = *(const float4*)(xj + i4);
    v4h ah, al, bh, bl;
    ah[0] = (_Float16)a.x; al[0] = (_Float16)(a.x - (float)ah[0]);
    ah[1] = (_Float16)a.y; al[1] = (_Float16)(a.y - (float)ah[1]);
    ah[2] = (_Float16)a.z; al[2] = (_Float16)(a.z - (float)ah[2]);
    ah[3] = (_Float16)a.w; al[3] = (_Float16)(a.w - (float)ah[3]);
    bh[0] = (_Float16)b.x; bl[0] = (_Float16)(b.x - (float)bh[0]);
    bh[1] = (_Float16)b.y; bl[1] = (_Float16)(b.y - (float)bh[1]);
    bh[2] = (_Float16)b.z; bl[2] = (_Float16)(b.z - (float)bh[2]);
    bh[3] = (_Float16)b.w; bl[3] = (_Float16)(b.w - (float)bh[3]);
    *(v4h*)(xiHi + i4) = ah;
    *(v4h*)(xiLo + i4) = al;
    *(v4h*)(xjHi + i4) = bh;
    *(v4h*)(xjLo + i4) = bl;
}

// ---------------------------------------------------------------------------
// Kernel 1: per-row norms, inverse norms, matched-pair similarity (pre-exp),
// computed in full f32 from the originals.
// ---------------------------------------------------------------------------
__global__ __launch_bounds__(128) void prep_kernel(
    const float* __restrict__ xi, const float* __restrict__ xj,
    float* __restrict__ invNi, float* __restrict__ invNj,
    float* __restrict__ smatch)
{
    const int row  = blockIdx.x;
    const int t    = threadIdx.x;
    const int lane = t & 31, wave = t >> 5;

    const float4 a = *(const float4*)(xi + (size_t)row * DIM + t * 4);
    const float4 b = *(const float4*)(xj + (size_t)row * DIM + t * 4);
    float ssi = a.x * a.x + a.y * a.y + a.z * a.z + a.w * a.w;
    float ssj = b.x * b.x + b.y * b.y + b.z * b.z + b.w * b.w;
    float dot = a.x * b.x + a.y * b.y + a.z * b.z + a.w * b.w;

#pragma unroll
    for (int m = 1; m < 32; m <<= 1) {
        ssi += __shfl_xor(ssi, m);
        ssj += __shfl_xor(ssj, m);
        dot += __shfl_xor(dot, m);
    }
    __shared__ float sA[4], sB[4], sC[4];
    if (lane == 0) { sA[wave] = ssi; sB[wave] = ssj; sC[wave] = dot; }
    __syncthreads();
    if (t == 0) {
        float si = sA[0] + sA[1] + sA[2] + sA[3];
        float sj = sB[0] + sB[1] + sB[2] + sB[3];
        float dd = sC[0] + sC[1] + sC[2] + sC[3];
        float ni = sqrtf(si), nj = sqrtf(sj);
        invNi[row]  = 1.0f / fmaxf(ni, 1e-16f);
        invNj[row]  = 1.0f / fmaxf(nj, 1e-16f);
        smatch[row] = dd / fmaxf(ni * nj, 1e-16f);   // log of matched-pair sim (TAU=1)
    }
}

// ---------------------------------------------------------------------------
// Kernel 2: fused  rowSumPartial[i] = sum_j exp( (xi_i . xj_j) * invni*invnj )
// Split-f16 GEMM via 3x v_wmma_f32_16x16x32_f16 per k-step; tiles staged with
// async global->LDS copies of the pre-converted f16 planes.
// Block: MB=64 rows; j loop over NB=64 cols; 8 waves: M-tile (w&3),
// N-tiles 2*(w>>2), 2*(w>>2)+1.
// ---------------------------------------------------------------------------
__global__ __launch_bounds__(256) void simsum_kernel(
    const _Float16* __restrict__ xiHi, const _Float16* __restrict__ xiLo,
    const _Float16* __restrict__ xjHi, const _Float16* __restrict__ xjLo,
    const float* __restrict__ invNi, const float* __restrict__ invNj,
    float* __restrict__ partial)
{
    __shared__ _Float16 lAhi[MB][KP], lAlo[MB][KP];
    __shared__ _Float16 lBhi[NB][KP], lBlo[NB][KP];
    __shared__ float    redbuf[MB][2];

    const int t    = threadIdx.x;
    const int lane = t & 31, wave = t >> 5;
    const int g    = lane >> 4;       // half-wave group (0/1)
    const int ln   = lane & 15;
    const int mt   = wave & 3;        // M tile index within block
    const int ntb  = (wave >> 2) * 2; // first N tile index
    const int rowBase = blockIdx.x * MB;

    // per-lane inverse row norms (C layout: VGPR v -> M = v + 8g)
    float invni_r[8];
#pragma unroll
    for (int v = 0; v < 8; ++v)
        invni_r[v] = invNi[rowBase + mt * 16 + v + 8 * g];

    float rs[8] = {0.f, 0.f, 0.f, 0.f, 0.f, 0.f, 0.f, 0.f};

    const int jbBeg = blockIdx.y * (N_ROWS / NB / JSPLIT);
    const int jbEnd = jbBeg + (N_ROWS / NB / JSPLIT);

    for (int jb = jbBeg; jb < jbEnd; ++jb) {
        const int j0 = jb * NB;
        v8f acc0 = {};
        v8f acc1 = {};

        const size_t aBase = (size_t)rowBase * DIM;
        const size_t bBase = (size_t)j0 * DIM;

        for (int kc = 0; kc < DIM / KC; ++kc) {
            const int kb0 = kc * KC;
            // stage 4 tiles (64 x 64 halves each) via async copies:
            // 512 16B-segments per tile, 2 per thread per tile.
#pragma unroll
            for (int it = 0; it < 2; ++it) {
                const int e = it * 256 + t;
                const int r = e >> 3;          // row within tile
                const int c = (e & 7) << 3;    // half offset, multiple of 8
                const size_t ga = aBase + (size_t)r * DIM + kb0 + c;
                const size_t gb = bBase + (size_t)r * DIM + kb0 + c;
                async_copy16(&lAhi[r][c], xiHi + ga);
                async_copy16(&lAlo[r][c], xiLo + ga);
                async_copy16(&lBhi[r][c], xjHi + gb);
                async_copy16(&lBlo[r][c], xjLo + gb);
            }
            wait_async0();
            __syncthreads();

#pragma unroll
            for (int ks = 0; ks < KC / 32; ++ks) {
                const int kb = ks * 32;
                // A frag (16x32): lane row = ln,
                // halves h -> K = kb + 16g + h
                const int arow = mt * 16 + ln;
                v8h a0  = *(const v8h*)&lAhi[arow][kb + g * 16];
                v8h a1  = *(const v8h*)&lAhi[arow][kb + g * 16 + 8];
                v8h a0l = *(const v8h*)&lAlo[arow][kb + g * 16];
                v8h a1l = *(const v8h*)&lAlo[arow][kb + g * 16 + 8];
                v16h Ahi = combine16(a0, a1);
                v16h Alo = combine16(a0l, a1l);

                // B frag (32x16): lane col = ln, halves h -> K = kb + 16g + h
                {
                    const int bcol = (ntb + 0) * 16 + ln;
                    v8h b0  = *(const v8h*)&lBhi[bcol][kb + g * 16];
                    v8h b1  = *(const v8h*)&lBhi[bcol][kb + g * 16 + 8];
                    v8h b0l = *(const v8h*)&lBlo[bcol][kb + g * 16];
                    v8h b1l = *(const v8h*)&lBlo[bcol][kb + g * 16 + 8];
                    v16h Bhi = combine16(b0, b1);
                    v16h Blo = combine16(b0l, b1l);
                    acc0 = __builtin_amdgcn_wmma_f32_16x16x32_f16(false, Ahi, false, Bhi, (short)0, acc0, false, false);
                    acc0 = __builtin_amdgcn_wmma_f32_16x16x32_f16(false, Ahi, false, Blo, (short)0, acc0, false, false);
                    acc0 = __builtin_amdgcn_wmma_f32_16x16x32_f16(false, Alo, false, Bhi, (short)0, acc0, false, false);
                }
                {
                    const int bcol = (ntb + 1) * 16 + ln;
                    v8h b0  = *(const v8h*)&lBhi[bcol][kb + g * 16];
                    v8h b1  = *(const v8h*)&lBhi[bcol][kb + g * 16 + 8];
                    v8h b0l = *(const v8h*)&lBlo[bcol][kb + g * 16];
                    v8h b1l = *(const v8h*)&lBlo[bcol][kb + g * 16 + 8];
                    v16h Bhi = combine16(b0, b1);
                    v16h Blo = combine16(b0l, b1l);
                    acc1 = __builtin_amdgcn_wmma_f32_16x16x32_f16(false, Ahi, false, Bhi, (short)0, acc1, false, false);
                    acc1 = __builtin_amdgcn_wmma_f32_16x16x32_f16(false, Ahi, false, Blo, (short)0, acc1, false, false);
                    acc1 = __builtin_amdgcn_wmma_f32_16x16x32_f16(false, Alo, false, Bhi, (short)0, acc1, false, false);
                }
            }
            __syncthreads();   // protect LDS tiles before next async stage
        }

        // epilogue: scale, exp, accumulate per-lane row partial sums
        const float invnj0 = invNj[j0 + (ntb + 0) * 16 + ln];
        const float invnj1 = invNj[j0 + (ntb + 1) * 16 + ln];
#pragma unroll
        for (int v = 0; v < 8; ++v) {
            rs[v] += __expf(acc0[v] * (invni_r[v] * invnj0));
            rs[v] += __expf(acc1[v] * (invni_r[v] * invnj1));
        }
    }

    // reduce 16 lanes per row, combine the 2 waves sharing an M tile,
    // write deterministic per-(block, j-split) partial sums.
#pragma unroll
    for (int v = 0; v < 8; ++v) {
        float r = rs[v];
        r += __shfl_xor(r, 1);
        r += __shfl_xor(r, 2);
        r += __shfl_xor(r, 4);
        r += __shfl_xor(r, 8);
        if (ln == 0) redbuf[mt * 16 + v + 8 * g][wave >> 2] = r;
    }
    __syncthreads();
    if (t < MB) {
        float tot = redbuf[t][0] + redbuf[t][1];
        partial[(size_t)blockIdx.y * N_ROWS + rowBase + t] = tot;
    }
}

// ---------------------------------------------------------------------------
// Kernel 3: loss = mean_i [ log(sum_splits partial - e) - smatch_i ]
// ---------------------------------------------------------------------------
__global__ __launch_bounds__(256) void finalize_kernel(
    const float* __restrict__ partial, const float* __restrict__ smatch,
    float* __restrict__ out)
{
    const int t = threadIdx.x;
    float acc = 0.f;
    for (int i = t; i < N_ROWS; i += 256) {
        float rsum = partial[i] + partial[N_ROWS + i] + partial[2 * N_ROWS + i]
                   + partial[3 * N_ROWS + i] - EULER_E;
        acc += logf(rsum) - smatch[i];
    }
#pragma unroll
    for (int m = 1; m < 32; m <<= 1) acc += __shfl_xor(acc, m);
    __shared__ float s[8];
    const int lane = t & 31, wave = t >> 5;
    if (lane == 0) s[wave] = acc;
    __syncthreads();
    if (t == 0) {
        float tot = 0.f;
        for (int w = 0; w < 8; ++w) tot += s[w];
        out[0] = tot / (float)N_ROWS;
    }
}

extern "C" void kernel_launch(void* const* d_in, const int* in_sizes, int n_in,
                              void* d_out, int out_size, void* d_ws, size_t ws_size,
                              hipStream_t stream) {
    const float* xi = (const float*)d_in[0];
    const float* xj = (const float*)d_in[1];

    float* ws      = (float*)d_ws;
    float* invNi   = ws;
    float* invNj   = ws + N_ROWS;
    float* smatch  = ws + 2 * N_ROWS;
    float* partial = ws + 3 * N_ROWS;            // JSPLIT * N_ROWS floats

    _Float16* f16b = (_Float16*)(ws + (3 + JSPLIT) * N_ROWS);  // 16B-aligned
    _Float16* xiHi = f16b;
    _Float16* xiLo = f16b + NMAT;
    _Float16* xjHi = f16b + 2 * NMAT;
    _Float16* xjLo = f16b + 3 * NMAT;

    cvt_kernel<<<(int)(NMAT / 1024), 256, 0, stream>>>(xi, xj, xiHi, xiLo, xjHi, xjLo);
    prep_kernel<<<N_ROWS, 128, 0, stream>>>(xi, xj, invNi, invNj, smatch);

    dim3 grid(N_ROWS / MB, JSPLIT);
    simsum_kernel<<<grid, 256, 0, stream>>>(xiHi, xiLo, xjHi, xjLo, invNi, invNj, partial);

    finalize_kernel<<<1, 256, 0, stream>>>(partial, smatch, (float*)d_out);
}
